// ColAttentionLayer_39788577030272
// MI455X (gfx1250) — compile-verified, matrix operand order checked by reading, not run on hardware
//
#include <hip/hip_runtime.h>
#include <hip/hip_bf16.h>

#define B_ 128
#define T_ 2048
#define D_ 512
#define H_ 512

typedef __attribute__((ext_vector_type(16))) __bf16 v16bf;
typedef __attribute__((ext_vector_type(8)))  float  v8f;

union FragBF { v16bf v; unsigned int u[8]; };

// pack two f32 -> packed bf16 (low half = a, high half = b)
__device__ __forceinline__ unsigned int pack2_bf16(float a, float b) {
#if __has_builtin(__builtin_amdgcn_cvt_pk_bf16_f32)
  typedef __attribute__((ext_vector_type(2))) __bf16 v2bf;
  union { v2bf v; unsigned int u; } r;
  r.v = __builtin_amdgcn_cvt_pk_bf16_f32(a, b);
  return r.u;
#elif __has_builtin(__builtin_amdgcn_perm)
  // single v_perm_b32: bytes {b3,b2,a3,a2} == {bf16(b), bf16(a)} (truncate)
  return __builtin_amdgcn_perm(__float_as_uint(b), __float_as_uint(a), 0x07060302u);
#else
  return (__float_as_uint(a) >> 16) | (__float_as_uint(b) & 0xFFFF0000u);
#endif
}

// ---------------- Kernel 1: col_t = relu(col_output @ W^T)  [B,H] ------------
__global__ void __launch_bounds__(256)
k_col_proj(const float* __restrict__ col, const float* __restrict__ W,
           float* __restrict__ colT) {
  __shared__ float lc[D_];
  const int b = blockIdx.x, tid = threadIdx.x;
  for (int i = tid; i < D_; i += 256) lc[i] = col[b * D_ + i];
  __syncthreads();
  for (int h = tid; h < H_; h += 256) {
    const float* wr = W + (size_t)h * D_;
    float acc = 0.f;
#pragma unroll 8
    for (int d = 0; d < D_; ++d) acc = fmaf(lc[d], wr[d], acc);
    colT[b * H_ + h] = fmaxf(acc, 0.f);
  }
}

// ---------------- Kernel 2: scores[b,t] = col_t[b] . relu(x[b,t] @ W^T) ------
#define K2_THREADS 512          // 16 waves of 32
#define HP 128                  // h rows staged per pass (4 passes)
#define WSTR 264                // u32 per LDS W row: 256 data + 8 pad (bank spread)

__global__ void __launch_bounds__(K2_THREADS)
k_scores(const float* __restrict__ x, const int* __restrict__ mask,
         const float* __restrict__ W, const float* __restrict__ colT,
         float* __restrict__ scores) {
  __shared__ __align__(16) unsigned int ldsW[HP * WSTR];  // ~132 KB bf16 W chunk
  __shared__ float ldsC[H_];                              // col_t (2 KB)

  const int b    = blockIdx.y;
  const int tseg = blockIdx.x * 512;          // this block owns 512 t-rows
  const int tid  = threadIdx.x;
  const int lane = tid & 31, wave = tid >> 5;
  const int halfsel = lane >> 4;              // selects the K-run (ISA A/B layout)
  const int nsub    = lane & 15;              // row (A) / column (B,C) index

  for (int i = tid; i < H_; i += K2_THREADS) ldsC[i] = colT[b * H_ + i];

  // each wave owns 32 t-rows: tile0 = rows [tbase,tbase+16), tile1 = +16
  const int tbase = tseg + wave * 32;
  const float* xrow0 = x + ((size_t)b * T_ + (size_t)(tbase + nsub)) * D_;
  const float* xrow1 = xrow0 + (size_t)16 * D_;

  float part[2][8];
#pragma unroll
  for (int tl = 0; tl < 2; ++tl)
#pragma unroll
    for (int r = 0; r < 8; ++r) part[tl][r] = 0.f;

  for (int hp = 0; hp < 4; ++hp) {
    __syncthreads();
    // stage W[hp*128 .. +128)[0..511] as bf16 into LDS
    const float* wsrc = W + (size_t)(hp * HP) * D_;
    for (int i = tid; i < HP * (D_ / 2); i += K2_THREADS) {
      const int row = i >> 8;                 // D_/2 == 256 u32 per row
      const int c   = i & 255;
      ldsW[row * WSTR + c] =
          pack2_bf16(wsrc[row * D_ + 2 * c], wsrc[row * D_ + 2 * c + 1]);
    }
    __syncthreads();

    v8f acc0[8], acc1[8];
#pragma unroll
    for (int ht = 0; ht < 8; ++ht)
#pragma unroll
      for (int r = 0; r < 8; ++r) { acc0[ht][r] = 0.f; acc1[ht][r] = 0.f; }

    // raw f32 staging registers; loads for kc+1 issued under the wmma loop
    float4 r0a, r0b, r0c, r0d, r1a, r1b, r1c, r1d;
    {
      const float* p1 = xrow0 + halfsel * 8;
      const float* p2 = xrow0 + 16 + halfsel * 8;
      const float* q1 = xrow1 + halfsel * 8;
      const float* q2 = xrow1 + 16 + halfsel * 8;
      r0a = *(const float4*)(p1); r0b = *(const float4*)(p1 + 4);
      r0c = *(const float4*)(p2); r0d = *(const float4*)(p2 + 4);
      r1a = *(const float4*)(q1); r1b = *(const float4*)(q1 + 4);
      r1c = *(const float4*)(q2); r1d = *(const float4*)(q2 + 4);
    }

    for (int kc = 0; kc < 16; ++kc) {         // K = 512 in 16 steps of 32
      const int k0 = kc * 32;

      // pack current raws into two 16x32 bf16 A-fragments (one v_perm each u32)
      FragBF a0, a1;
      a0.u[0] = pack2_bf16(r0a.x, r0a.y); a0.u[1] = pack2_bf16(r0a.z, r0a.w);
      a0.u[2] = pack2_bf16(r0b.x, r0b.y); a0.u[3] = pack2_bf16(r0b.z, r0b.w);
      a0.u[4] = pack2_bf16(r0c.x, r0c.y); a0.u[5] = pack2_bf16(r0c.z, r0c.w);
      a0.u[6] = pack2_bf16(r0d.x, r0d.y); a0.u[7] = pack2_bf16(r0d.z, r0d.w);
      a1.u[0] = pack2_bf16(r1a.x, r1a.y); a1.u[1] = pack2_bf16(r1a.z, r1a.w);
      a1.u[2] = pack2_bf16(r1b.x, r1b.y); a1.u[3] = pack2_bf16(r1b.z, r1b.w);
      a1.u[4] = pack2_bf16(r1c.x, r1c.y); a1.u[5] = pack2_bf16(r1c.z, r1c.w);
      a1.u[6] = pack2_bf16(r1d.x, r1d.y); a1.u[7] = pack2_bf16(r1d.z, r1d.w);

      // issue next kc's global loads early; they retire under the wmma loop
      if (kc < 15) {
        const int kn = k0 + 32;
        const float* p1 = xrow0 + kn + halfsel * 8;
        const float* p2 = xrow0 + kn + 16 + halfsel * 8;
        const float* q1 = xrow1 + kn + halfsel * 8;
        const float* q2 = xrow1 + kn + 16 + halfsel * 8;
        r0a = *(const float4*)(p1); r0b = *(const float4*)(p1 + 4);
        r0c = *(const float4*)(p2); r0d = *(const float4*)(p2 + 4);
        r1a = *(const float4*)(q1); r1b = *(const float4*)(q1 + 4);
        r1c = *(const float4*)(q2); r1d = *(const float4*)(q2 + 4);
      }

#pragma unroll
      for (int ht = 0; ht < 8; ++ht) {
        // B-fragment (W^T 32x16): lane holds column h = ht*16 + nsub.
        // One B fetch feeds TWO wmmas (tile0 + tile1).
        FragBF bf;
        const unsigned int* wrow = ldsW + (ht * 16 + nsub) * WSTR;
        const uint4 q0 = *(const uint4*)(wrow + (k0 >> 1) + halfsel * 4);
        const uint4 q1 = *(const uint4*)(wrow + (k0 >> 1) + 8 + halfsel * 4);
        bf.u[0] = q0.x; bf.u[1] = q0.y; bf.u[2] = q0.z; bf.u[3] = q0.w;
        bf.u[4] = q1.x; bf.u[5] = q1.y; bf.u[6] = q1.z; bf.u[7] = q1.w;

        acc0[ht] = __builtin_amdgcn_wmma_f32_16x16x32_bf16(
            false, a0.v, false, bf.v, (short)0, acc0[ht], false, false);
        acc1[ht] = __builtin_amdgcn_wmma_f32_16x16x32_bf16(
            false, a1.v, false, bf.v, (short)0, acc1[ht], false, false);
      }
    }

    // fuse relu + col_t dot on the C layout (VGPR r = row, lane%16 = col)
#pragma unroll
    for (int ht = 0; ht < 8; ++ht) {
      const float colv = ldsC[hp * HP + ht * 16 + nsub];
#pragma unroll
      for (int r = 0; r < 8; ++r) {
        part[0][r] = fmaf(fmaxf(acc0[ht][r], 0.f), colv, part[0][r]);
        part[1][r] = fmaf(fmaxf(acc1[ht][r], 0.f), colv, part[1][r]);
      }
    }
  }

  // reduce the 16 columns held within each 16-lane half; write masked scores
#pragma unroll
  for (int tl = 0; tl < 2; ++tl) {
#pragma unroll
    for (int r = 0; r < 8; ++r) {
      float p = part[tl][r];
      p += __shfl_xor(p, 1, 32);
      p += __shfl_xor(p, 2, 32);
      p += __shfl_xor(p, 4, 32);
      p += __shfl_xor(p, 8, 32);
      if (nsub == 0) {
        const int t = tbase + tl * 16 + r + halfsel * 8;
        scores[(size_t)b * T_ + t] = mask[b * T_ + t] ? p : 0.f;
      }
    }
  }
}

// ------- Kernel 3: softmax over T, weighted sum over x, concat outputs -------
__global__ void __launch_bounds__(512)
k_softmax_wsum(const float* __restrict__ scores, const float* __restrict__ x,
               const float* __restrict__ col, float* __restrict__ out) {
  __shared__ float lw[T_];     // exp weights (8 KB)
  __shared__ float red[16];
  const int b = blockIdx.x, tid = threadIdx.x;
  const int lane = tid & 31, wave = tid >> 5;

  float sv[4];
#pragma unroll
  for (int j = 0; j < 4; ++j) sv[j] = scores[(size_t)b * T_ + tid + j * 512];

  float m = fmaxf(fmaxf(sv[0], sv[1]), fmaxf(sv[2], sv[3]));
#pragma unroll
  for (int o = 1; o < 32; o <<= 1) m = fmaxf(m, __shfl_xor(m, o, 32));
  if (lane == 0) red[wave] = m;
  __syncthreads();
  if (tid == 0) {
    float mm = red[0];
    for (int i = 1; i < 16; ++i) mm = fmaxf(mm, red[i]);
    red[0] = mm;
  }
  __syncthreads();
  m = red[0];

  float lsum = 0.f;
#pragma unroll
  for (int j = 0; j < 4; ++j) {
    const float e = __expf(sv[j] - m);
    lw[tid + j * 512] = e;
    lsum += e;
  }
#pragma unroll
  for (int o = 1; o < 32; o <<= 1) lsum += __shfl_xor(lsum, o, 32);
  __syncthreads();                      // all reads of red[0] done before reuse
  if (lane == 0) red[wave] = lsum;
  __syncthreads();
  if (tid == 0) {
    float ss = 0.f;
    for (int i = 0; i < 16; ++i) ss += red[i];
    red[0] = ss;
  }
  __syncthreads();
  const float inv = 1.f / red[0];

  // out[d] = (1/L) * sum_t e[t] * x[b,t,d]; one d per thread, coalesced rows
  const int d = tid;
  const float* xp = x + (size_t)b * T_ * D_ + d;
  float acc = 0.f;
#pragma unroll 8
  for (int t = 0; t < T_; ++t) acc = fmaf(lw[t], xp[(size_t)t * D_], acc);
  acc *= inv;

  const float co = col[b * D_ + d];
  out[(size_t)b * (2 * D_) + d]        = co;    // cat[:, :D]  = col_output
  out[(size_t)b * (2 * D_) + D_ + d]   = acc;   // cat[:, D:]  = attention_output
  out[(size_t)B_ * (2 * D_) + (size_t)b * D_ + d] = acc;  // attention_output
}

// -----------------------------------------------------------------------------
extern "C" void kernel_launch(void* const* d_in, const int* in_sizes, int n_in,
                              void* d_out, int out_size, void* d_ws, size_t ws_size,
                              hipStream_t stream) {
  const float* col  = (const float*)d_in[0];   // [B, D]
  const float* x    = (const float*)d_in[1];   // [B, T, D]
  const int*   mask = (const int*)  d_in[2];   // [B, T]
  const float* W    = (const float*)d_in[3];   // [H, D]
  float* out = (float*)d_out;                  // [B,2D] ++ [B,D]

  float* ws_colT   = (float*)d_ws;             // B*H  f32
  float* ws_scores = ws_colT + B_ * H_;        // B*T  f32  (total 1.25 MB)

  k_col_proj<<<B_, 256, 0, stream>>>(col, W, ws_colT);
  k_scores<<<dim3(4, B_), K2_THREADS, 0, stream>>>(x, mask, W, ws_colT, ws_scores);
  k_softmax_wsum<<<B_, 512, 0, stream>>>(ws_scores, x, col, out);
}
